// GRUAttn_KeyCNN_AllKeys_29248727286154
// MI455X (gfx1250) — matrix operand order repaired and checked by reading, not compiled, not run on hardware
//
#include <hip/hip_runtime.h>
#include <hip/hip_bf16.h>
#include <math.h>

// ---------------------------------------------------------------------------
// Problem constants (fixed by the reference):
//   B=32, S=160, K=44, E=300, H=256, 3H=768
// ---------------------------------------------------------------------------
#define BB   32
#define SS   160
#define KKT  44
#define EE   300
#define HH   256
#define G3   768          // 3*H
#define XPK  320          // E padded to multiple of 32 (key GRU input width)
#define XPE  576          // (E+H)=556 padded to multiple of 32 (main GRU input)
#define XA_STRIDE 584     // LDS row stride for x tile (pad -> conflict-free)
#define H_STRIDE  264     // LDS row stride for h tile

typedef __bf16 bf16;
typedef __attribute__((ext_vector_type(16))) __bf16 v16bf;
typedef __attribute__((ext_vector_type(8)))  float  v8f;
typedef __attribute__((ext_vector_type(4)))  unsigned uint4v;
typedef __attribute__((ext_vector_type(2)))  unsigned uint2v;
typedef __attribute__((ext_vector_type(4)))  float    float4v;

// A-fragment (16x32 bf16, M x K): lane m=lane&15 (hi=lane>>4) holds
// K = [hi*8 .. hi*8+7] and [16+hi*8 .. 16+hi*8+7] of its row.
__device__ __forceinline__ v16bf lds_a_frag(const bf16* p) {
  union { v16bf v; uint4v q[2]; } u;
  u.q[0] = *(const uint4v*)p;
  u.q[1] = *(const uint4v*)(p + 16);
  return u.v;
}
// B-fragment (32x16 bf16, K x N) with B[k,n] = W[n,k]: lane n=lane&15 holds
// K = [hi*8 .. hi*8+7] and [16+hi*8 .. 16+hi*8+7] of weight row n.
__device__ __forceinline__ v16bf glb_b_frag(const bf16* p) {
  union { v16bf v; uint4v q[2]; } u;
  u.q[0] = *(const uint4v*)p;
  u.q[1] = *(const uint4v*)(p + 16);
  return u.v;
}
__device__ __forceinline__ v8f wmma_bf16(v16bf a, v16bf b, v8f c) {
  return __builtin_amdgcn_wmma_f32_16x16x32_bf16(false, a, false, b,
                                                 (short)0, c, false, false);
}
__device__ __forceinline__ float sigm(float x) { return 1.f / (1.f + __expf(-x)); }

// CDNA5 async global->LDS copy (ASYNCcnt path). vdst = 32-bit LDS offset,
// vaddr = 64-bit global address.
__device__ __forceinline__ void async_copy_b128(unsigned lds_off, const void* g) {
  asm volatile("global_load_async_to_lds_b128 %0, %1, off"
               :: "v"(lds_off), "v"(g) : "memory");
}
__device__ __forceinline__ void wait_async0() {
  asm volatile("s_wait_asynccnt 0x0" ::: "memory");
}

// ---------------------------------------------------------------------------
// Persistent per-tile GRU. One workgroup = 16 sequences, 8 waves (wave32).
// Wave w owns output columns [32w,32w+32) in each of the 3 gate regions
// (2 N-tiles of 16 per region). h state lives in LDS as bf16; x_t is staged
// (async-copied, or gathered from the embedding table) into LDS each step.
// XLD (padded input width) is a template constant so the z/n weight planes
// become compile-time load offsets (fit the 24-bit global-load immediate).
// B-fragments are ping/pong double-buffered (kc+=2) so the global loads of
// chunk k+1 issue while the WMMAs of chunk k execute, with no copy moves.
// ---------------------------------------------------------------------------
template <int XLD>
__global__ __launch_bounds__(256) void gru_wmma_kernel(
    const bf16* __restrict__ xrows,   // pregathered [nseq*T, XLD] bf16, or null
    const int*  __restrict__ keys,    // gather mode: token ids [nseq*T]
    const float* __restrict__ emb,    // gather mode: [V, EE] f32
    const bf16* __restrict__ Wih,     // [768, XLD] bf16
    const bf16* __restrict__ Whh,     // [768, 256] bf16
    const float* __restrict__ bih, const float* __restrict__ bhh,
    int nseq, int T, int reverse,
    float* __restrict__ hT, int hT_add, int hT_stride, int hT_coloff,
    float* __restrict__ seq_out, int seq_stride, int seq_coloff)
{
  constexpr int KX32 = XLD / 32;                // even (10 or 18)
  constexpr int KH32 = HH / 32;                 // 8
  constexpr size_t OZI = (size_t)HH * XLD;      // z-plane offset in Wih (elems)
  constexpr size_t ONI = (size_t)2 * HH * XLD;  // n-plane offset in Wih
  constexpr size_t OZH = (size_t)HH * HH;       // z-plane offset in Whh
  constexpr size_t ONH = (size_t)2 * HH * HH;   // n-plane offset in Whh

  __shared__ __align__(16) bf16 xa[16 * XA_STRIDE];
  __shared__ __align__(16) bf16 hb[16 * H_STRIDE];

  const int tid  = threadIdx.x;
  const int lane = tid & 31;
  const int wv   = tid >> 5;           // wave 0..7
  const int hi   = lane >> 4;          // 0/1 half-wave
  const int ml   = lane & 15;          // m (A/C rows) or n (B/C cols) local
  const int sb   = blockIdx.x * 16;    // sequence tile base
  const int cbase = wv * 32;           // this wave's column base per region

  for (int i = tid; i < 16 * H_STRIDE; i += 256) hb[i] = (bf16)0.f;

  // Per-thread weight row bases (z/n planes reached via constant offsets).
  const bf16 *wiP[2], *whP[2];
  float bR[2], bZ[2], bIN[2], bHN[2];
#pragma unroll
  for (int j = 0; j < 2; ++j) {
    int c  = cbase + j * 16 + ml;
    wiP[j] = Wih + (size_t)c * XLD;
    whP[j] = Whh + (size_t)c * HH;
    bR[j]  = bih[c] + bhh[c];
    bZ[j]  = bih[HH + c] + bhh[HH + c];
    bIN[j] = bih[2 * HH + c];
    bHN[j] = bhh[2 * HH + c];
  }

  // Staging geometry: 16 threads per row.
  const int srow = tid >> 4, sli = tid & 15;
  const int sseq = sb + srow;

  for (int t = 0; t < T; ++t) {
    const int tm = reverse ? (T - 1 - t) : t;

    // ---- stage x_t into LDS ----
    if (xrows) {
      // async global->LDS, 16B chunks (row is 16B-aligned: XLD*2 % 16 == 0)
      if (sseq < nseq) {
        const bf16* src = xrows + ((size_t)sseq * T + tm) * XLD;
        unsigned dst = (unsigned)(uintptr_t)(xa + srow * XA_STRIDE);
        for (int c = sli; c < XLD / 8; c += 16)
          async_copy_b128(dst + c * 16, src + c * 8);
      } else {
        uint2v zz = {0u, 0u};
        for (int e = sli * 4; e < XLD; e += 64)
          *(uint2v*)(xa + srow * XA_STRIDE + e) = zz;
      }
      wait_async0();
    } else {
      // fused embedding gather + f32->bf16 (emb rows are 16B aligned)
      const float* er = nullptr;
      if (sseq < nseq)
        er = emb + (size_t)keys[(size_t)sseq * T + tm] * EE;
      for (int e = sli * 4; e < XLD; e += 64) {
        uint2v pk = {0u, 0u};
        if (er && e < EE) {
          float4v f = *(const float4v*)(er + e);
          union { bf16 h[4]; uint2v u; } cv;
          cv.h[0] = (bf16)f.x; cv.h[1] = (bf16)f.y;
          cv.h[2] = (bf16)f.z; cv.h[3] = (bf16)f.w;
          pk = cv.u;
        }
        *(uint2v*)(xa + srow * XA_STRIDE + e) = pk;
      }
    }
    __syncthreads();

    const v8f vz = {0.f, 0.f, 0.f, 0.f, 0.f, 0.f, 0.f, 0.f};
    v8f accR[2]  = {vz, vz}, accZ[2] = {vz, vz};
    v8f accIN[2] = {vz, vz}, accHN[2] = {vz, vz};

    // ---- gi: x_t @ Wih^T (ping/pong double-buffered B fragments) ----
    {
      const bf16* arow = xa + ml * XA_STRIDE + hi * 8;
      v16bf bp[6], bq[6];
      auto loadTo = [&](v16bf (&b)[6], int kc) {
        int ko = kc * 32 + hi * 8;
        b[0] = glb_b_frag(wiP[0] + ko);        b[1] = glb_b_frag(wiP[1] + ko);
        b[2] = glb_b_frag(wiP[0] + OZI + ko);  b[3] = glb_b_frag(wiP[1] + OZI + ko);
        b[4] = glb_b_frag(wiP[0] + ONI + ko);  b[5] = glb_b_frag(wiP[1] + ONI + ko);
      };
      loadTo(bp, 0);
#pragma unroll 1
      for (int kc = 0; kc < KX32; kc += 2) {
        v16bf a0 = lds_a_frag(arow + kc * 32);
        loadTo(bq, kc + 1);
        accR[0]  = wmma_bf16(a0, bp[0], accR[0]);
        accR[1]  = wmma_bf16(a0, bp[1], accR[1]);
        accZ[0]  = wmma_bf16(a0, bp[2], accZ[0]);
        accZ[1]  = wmma_bf16(a0, bp[3], accZ[1]);
        accIN[0] = wmma_bf16(a0, bp[4], accIN[0]);
        accIN[1] = wmma_bf16(a0, bp[5], accIN[1]);
        v16bf a1 = lds_a_frag(arow + (kc + 1) * 32);
        if (kc + 2 < KX32) loadTo(bp, kc + 2);
        accR[0]  = wmma_bf16(a1, bq[0], accR[0]);
        accR[1]  = wmma_bf16(a1, bq[1], accR[1]);
        accZ[0]  = wmma_bf16(a1, bq[2], accZ[0]);
        accZ[1]  = wmma_bf16(a1, bq[3], accZ[1]);
        accIN[0] = wmma_bf16(a1, bq[4], accIN[0]);
        accIN[1] = wmma_bf16(a1, bq[5], accIN[1]);
      }
    }
    // ---- gh: h @ Whh^T (same ping/pong scheme) ----
    {
      const bf16* hrow = hb + ml * H_STRIDE + hi * 8;
      v16bf bp[6], bq[6];
      auto loadTo = [&](v16bf (&b)[6], int kc) {
        int ko = kc * 32 + hi * 8;
        b[0] = glb_b_frag(whP[0] + ko);        b[1] = glb_b_frag(whP[1] + ko);
        b[2] = glb_b_frag(whP[0] + OZH + ko);  b[3] = glb_b_frag(whP[1] + OZH + ko);
        b[4] = glb_b_frag(whP[0] + ONH + ko);  b[5] = glb_b_frag(whP[1] + ONH + ko);
      };
      loadTo(bp, 0);
#pragma unroll 1
      for (int kc = 0; kc < KH32; kc += 2) {
        v16bf a0 = lds_a_frag(hrow + kc * 32);
        loadTo(bq, kc + 1);
        accR[0]  = wmma_bf16(a0, bp[0], accR[0]);
        accR[1]  = wmma_bf16(a0, bp[1], accR[1]);
        accZ[0]  = wmma_bf16(a0, bp[2], accZ[0]);
        accZ[1]  = wmma_bf16(a0, bp[3], accZ[1]);
        accHN[0] = wmma_bf16(a0, bp[4], accHN[0]);
        accHN[1] = wmma_bf16(a0, bp[5], accHN[1]);
        v16bf a1 = lds_a_frag(hrow + (kc + 1) * 32);
        if (kc + 2 < KH32) loadTo(bp, kc + 2);
        accR[0]  = wmma_bf16(a1, bq[0], accR[0]);
        accR[1]  = wmma_bf16(a1, bq[1], accR[1]);
        accZ[0]  = wmma_bf16(a1, bq[2], accZ[0]);
        accZ[1]  = wmma_bf16(a1, bq[3], accZ[1]);
        accHN[0] = wmma_bf16(a1, bq[4], accHN[0]);
        accHN[1] = wmma_bf16(a1, bq[5], accHN[1]);
      }
    }
    // h_prev at this wave's C-layout coordinates (lane: n=ml, rows hi*8+i)
    float hprev[2][8];
#pragma unroll
    for (int j = 0; j < 2; ++j) {
      int c = cbase + j * 16 + ml;
#pragma unroll
      for (int i = 0; i < 8; ++i)
        hprev[j][i] = (float)hb[(hi * 8 + i) * H_STRIDE + c];
    }
    __syncthreads();  // all reads of xa/hb complete before rewrite

    // ---- gates (pure elementwise in the WMMA C layout) + h update ----
#pragma unroll
    for (int j = 0; j < 2; ++j) {
      int c = cbase + j * 16 + ml;
#pragma unroll
      for (int i = 0; i < 8; ++i) {
        float r  = sigm(accR[j][i] + bR[j]);
        float z  = sigm(accZ[j][i] + bZ[j]);
        float n  = tanhf(accIN[j][i] + bIN[j] + r * (accHN[j][i] + bHN[j]));
        float hn = (1.f - z) * n + z * hprev[j][i];
        int m = hi * 8 + i;
        hb[m * H_STRIDE + c] = (bf16)hn;
        if (seq_out) {
          int s = sb + m;
          if (s < nseq)
            seq_out[((size_t)s * T + tm) * seq_stride + seq_coloff + c] = hn;
        }
      }
    }
  }
  __syncthreads();
  if (hT) {
    for (int idx = tid; idx < 16 * HH; idx += 256) {
      int m = idx >> 8, c = idx & (HH - 1);
      int s = sb + m;
      if (s < nseq) {
        float v  = (float)hb[m * H_STRIDE + c];
        float* p = hT + (size_t)s * hT_stride + hT_coloff + c;
        if (hT_add) *p += v; else *p = v;
      }
    }
  }
}

// ---------------------------------------------------------------------------
// f32 weights -> bf16 with K-dim zero padding.
// ---------------------------------------------------------------------------
__global__ void convert_pad_bf16(const float* __restrict__ W,
                                 bf16* __restrict__ out,
                                 int N, int Kraw, int Kpad) {
  int idx = blockIdx.x * 256 + threadIdx.x;
  if (idx >= N * Kpad) return;
  int n = idx / Kpad, k = idx - n * Kpad;
  out[idx] = (k < Kraw) ? (bf16)W[(size_t)n * Kraw + k] : (bf16)0.f;
}

// rows of [ emb[tok] (300) | kh_row (256) | zero pad (20) ] as bf16.
__global__ void concat_x_bf16(const int* __restrict__ tok,
                              const float* __restrict__ emb,
                              const float* __restrict__ kh,
                              bf16* __restrict__ out, int rows) {
  int idx = blockIdx.x * 256 + threadIdx.x;
  if (idx >= rows * XPE) return;
  int r = idx / XPE, e = idx - r * XPE;
  float v = 0.f;
  if (e < EE)            v = emb[(size_t)tok[r] * EE + e];
  else if (e < EE + HH)  v = kh[(size_t)r * HH + (e - EE)];
  out[idx] = (bf16)v;
}

// ---------------------------------------------------------------------------
// Attention, algebraically reduced:
//   v[b]  = attn_W^T r[b]   (energies = sc . v + attn_b . r[b])
//   Mr[b] = M r[b]          (o = sum_s alpha_s (sc_s . Mr) + b0)
// ---------------------------------------------------------------------------
__global__ __launch_bounds__(256) void attn_prep(
    const float* __restrict__ r, const float* __restrict__ attn_W,
    const float* __restrict__ attn_b, const float* __restrict__ M,
    float* __restrict__ v, float* __restrict__ Mr, float* __restrict__ bdot) {
  int b = blockIdx.x;
  const float* rb = r + (size_t)b * 2 * HH;
  for (int e = threadIdx.x; e < 2 * HH; e += 256) {
    float s = 0.f;
    for (int d = 0; d < 2 * HH; ++d) s += attn_W[(size_t)d * 2 * HH + e] * rb[d];
    v[(size_t)b * 2 * HH + e] = s;
  }
  for (int d = threadIdx.x; d < 2 * HH; d += 256) {
    float s = 0.f;
    for (int e = 0; e < 2 * HH; ++e) s += M[(size_t)d * 2 * HH + e] * rb[e];
    Mr[(size_t)b * 2 * HH + d] = s;
  }
  if (threadIdx.x == 0) {
    float s = 0.f;
    for (int d = 0; d < 2 * HH; ++d) s += attn_b[d] * rb[d];
    bdot[b] = s;
  }
}

__global__ __launch_bounds__(256) void attn_out_kernel(
    const float* __restrict__ sc, const float* __restrict__ v,
    const float* __restrict__ Mr, const float* __restrict__ bdot,
    const float* __restrict__ mask, const float* __restrict__ bsc,
    float* __restrict__ out) {
  int b = blockIdx.x, tid = threadIdx.x;
  __shared__ float en[SS], wsm[SS], red[256];
  const float* vb = v  + (size_t)b * 2 * HH;
  const float* mb = Mr + (size_t)b * 2 * HH;
  for (int s = tid; s < SS; s += 256) {
    const float* row = sc + ((size_t)b * SS + s) * 2 * HH;
    float e = 0.f, w = 0.f;
    for (int d = 0; d < 2 * HH; ++d) { e += row[d] * vb[d]; w += row[d] * mb[d]; }
    en[s] = e + bdot[b]; wsm[s] = w;
  }
  __syncthreads();
  float lm = -3.4e38f;
  for (int s = tid; s < SS; s += 256) lm = fmaxf(lm, en[s]);
  red[tid] = lm; __syncthreads();
  for (int st = 128; st; st >>= 1) { if (tid < st) red[tid] = fmaxf(red[tid], red[tid + st]); __syncthreads(); }
  float mx = red[0]; __syncthreads();
  float ls = 0.f;
  for (int s = tid; s < SS; s += 256) ls += __expf(en[s] - mx);
  red[tid] = ls; __syncthreads();
  for (int st = 128; st; st >>= 1) { if (tid < st) red[tid] += red[tid + st]; __syncthreads(); }
  float inv = 1.f / red[0]; __syncthreads();
  float lo = 0.f;
  for (int s = tid; s < SS; s += 256)
    lo += __expf(en[s] - mx) * inv * mask[(size_t)b * SS + s] * wsm[s];
  red[tid] = lo; __syncthreads();
  for (int st = 128; st; st >>= 1) { if (tid < st) red[tid] += red[tid + st]; __syncthreads(); }
  if (tid == 0) out[b] = red[0] + bsc[0];
}

// ---------------------------------------------------------------------------
static inline size_t al256(size_t x) { return (x + 255) & ~(size_t)255; }

extern "C" void kernel_launch(void* const* d_in, const int* in_sizes, int n_in,
                              void* d_out, int out_size, void* d_ws, size_t ws_size,
                              hipStream_t stream) {
  (void)in_sizes; (void)n_in; (void)out_size; (void)ws_size;
  const int*   x1     = (const int*)d_in[0];
  const int*   x2     = (const int*)d_in[1];
  const int*   keys_c = (const int*)d_in[2];
  const int*   keys_r = (const int*)d_in[3];
  const float* x1mask = (const float*)d_in[4];
  const float* emb    = (const float*)d_in[5];
  const float* kWif = (const float*)d_in[6],  *kWhf = (const float*)d_in[7];
  const float* kbif = (const float*)d_in[8],  *kbhf = (const float*)d_in[9];
  const float* kWib = (const float*)d_in[10], *kWhb = (const float*)d_in[11];
  const float* kbib = (const float*)d_in[12], *kbhb = (const float*)d_in[13];
  const float* eWif = (const float*)d_in[14], *eWhf = (const float*)d_in[15];
  const float* ebif = (const float*)d_in[16], *ebhf = (const float*)d_in[17];
  const float* eWib = (const float*)d_in[18], *eWhb = (const float*)d_in[19];
  const float* ebib = (const float*)d_in[20], *ebhb = (const float*)d_in[21];
  const float* attn_W = (const float*)d_in[22];
  const float* attn_b = (const float*)d_in[23];
  const float* Mmat   = (const float*)d_in[24];
  const float* bscal  = (const float*)d_in[25];

  char* base = (char*)d_ws; size_t off = 0;
  auto carve = [&](size_t bytes) { void* p = base + off; off += al256(bytes); return p; };

  bf16* kWif_bf = (bf16*)carve((size_t)G3 * XPK * 2);
  bf16* kWhf_bf = (bf16*)carve((size_t)G3 * HH * 2);
  bf16* kWib_bf = (bf16*)carve((size_t)G3 * XPK * 2);
  bf16* kWhb_bf = (bf16*)carve((size_t)G3 * HH * 2);
  bf16* eWif_bf = (bf16*)carve((size_t)G3 * XPE * 2);
  bf16* eWhf_bf = (bf16*)carve((size_t)G3 * HH * 2);
  bf16* eWib_bf = (bf16*)carve((size_t)G3 * XPE * 2);
  bf16* eWhb_bf = (bf16*)carve((size_t)G3 * HH * 2);
  float* kc   = (float*)carve((size_t)BB * SS * HH * 4);
  float* kr   = (float*)carve((size_t)BB * SS * HH * 4);
  bf16*  x1e  = (bf16*)carve((size_t)BB * SS * XPE * 2);
  bf16*  x2e  = (bf16*)carve((size_t)BB * SS * XPE * 2);
  float* sc   = (float*)carve((size_t)BB * SS * 2 * HH * 4);
  float* rbuf = (float*)carve((size_t)BB * 2 * HH * 4);
  float* vb   = (float*)carve((size_t)BB * 2 * HH * 4);
  float* Mrb  = (float*)carve((size_t)BB * 2 * HH * 4);
  float* bdot = (float*)carve((size_t)BB * 4);

  auto conv = [&](const float* W, bf16* o, int N, int Kr, int Kp) {
    int total = N * Kp;
    convert_pad_bf16<<<(total + 255) / 256, 256, 0, stream>>>(W, o, N, Kr, Kp);
  };
  conv(kWif, kWif_bf, G3, EE, XPK);       conv(kWhf, kWhf_bf, G3, HH, HH);
  conv(kWib, kWib_bf, G3, EE, XPK);       conv(kWhb, kWhb_bf, G3, HH, HH);
  conv(eWif, eWif_bf, G3, EE + HH, XPE);  conv(eWhf, eWhf_bf, G3, HH, HH);
  conv(eWib, eWib_bf, G3, EE + HH, XPE);  conv(eWhb, eWhb_bf, G3, HH, HH);

  const int nseqK = BB * SS;  // 5120 key sequences
  dim3 gK(nseqK / 16);
  // Key BiGRU over keys_c -> kc = hf + hb
  gru_wmma_kernel<XPK><<<gK, 256, 0, stream>>>(nullptr, keys_c, emb,
      kWif_bf, kWhf_bf, kbif, kbhf, nseqK, KKT, 0, kc, 0, HH, 0, nullptr, 0, 0);
  gru_wmma_kernel<XPK><<<gK, 256, 0, stream>>>(nullptr, keys_c, emb,
      kWib_bf, kWhb_bf, kbib, kbhb, nseqK, KKT, 1, kc, 1, HH, 0, nullptr, 0, 0);
  // Key BiGRU over keys_r -> kr
  gru_wmma_kernel<XPK><<<gK, 256, 0, stream>>>(nullptr, keys_r, emb,
      kWif_bf, kWhf_bf, kbif, kbhf, nseqK, KKT, 0, kr, 0, HH, 0, nullptr, 0, 0);
  gru_wmma_kernel<XPK><<<gK, 256, 0, stream>>>(nullptr, keys_r, emb,
      kWib_bf, kWhb_bf, kbib, kbhb, nseqK, KKT, 1, kr, 1, HH, 0, nullptr, 0, 0);

  {
    int total = nseqK * XPE;
    concat_x_bf16<<<(total + 255) / 256, 256, 0, stream>>>(x1, emb, kc, x1e, nseqK);
    concat_x_bf16<<<(total + 255) / 256, 256, 0, stream>>>(x2, emb, kr, x2e, nseqK);
  }

  // Encoder BiGRU over x1e -> sc (fwd cols [0,256), bwd cols [256,512))
  gru_wmma_kernel<XPE><<<BB / 16, 256, 0, stream>>>(x1e, nullptr, nullptr,
      eWif_bf, eWhf_bf, ebif, ebhf, BB, SS, 0, nullptr, 0, 0, 0, sc, 2 * HH, 0);
  gru_wmma_kernel<XPE><<<BB / 16, 256, 0, stream>>>(x1e, nullptr, nullptr,
      eWib_bf, eWhb_bf, ebib, ebhb, BB, SS, 1, nullptr, 0, 0, 0, sc, 2 * HH, HH);
  // Encoder BiGRU over x2e -> r = [r1f | r1b]
  gru_wmma_kernel<XPE><<<BB / 16, 256, 0, stream>>>(x2e, nullptr, nullptr,
      eWif_bf, eWhf_bf, ebif, ebhf, BB, SS, 0, rbuf, 0, 2 * HH, 0, nullptr, 0, 0);
  gru_wmma_kernel<XPE><<<BB / 16, 256, 0, stream>>>(x2e, nullptr, nullptr,
      eWib_bf, eWhb_bf, ebib, ebhb, BB, SS, 1, rbuf, 0, 2 * HH, HH, nullptr, 0, 0);

  attn_prep<<<BB, 256, 0, stream>>>(rbuf, attn_W, attn_b, Mmat, vb, Mrb, bdot);
  attn_out_kernel<<<BB, 256, 0, stream>>>(sc, vb, Mrb, bdot, x1mask, bscal,
                                          (float*)d_out);
}
